// Grid_Attention_69973607187056
// MI455X (gfx1250) — matrix operand order repaired and checked by reading, not compiled, not run on hardware
//
#include <hip/hip_runtime.h>

typedef _Float16 h2   __attribute__((ext_vector_type(2)));
typedef _Float16 h4   __attribute__((ext_vector_type(4)));
typedef _Float16 h8   __attribute__((ext_vector_type(8)));
typedef _Float16 h16  __attribute__((ext_vector_type(16)));
typedef float    f8x  __attribute__((ext_vector_type(8)));
typedef int      i4v  __attribute__((ext_vector_type(4)));

#define DMODEL 512
#define NTOK   49
#define HEADS  16
#define DHEAD  32
#define NWIN   4096            // B*X*Y = 4*32*32
#define MTOT   200704          // NWIN*NTOK

// d_ws layout (in _Float16 elements). Weights stored TRANSPOSED: WT[n][k].
#define WQ_OFF 0u
#define WK_OFF 262144u
#define WV_OFF 524288u
#define Q_OFF  786432u
#define K_OFF  (Q_OFF + 102760448u)
#define V_OFF  (K_OFF + 102760448u)

// ---- CDNA5 async global->LDS copy (ASYNCcnt-tracked), with sync fallback ----
#if __has_builtin(__builtin_amdgcn_global_load_async_to_lds_b128)
#define ASYNC_LDS 1
#else
#define ASYNC_LDS 0
#endif

typedef __attribute__((address_space(1))) i4v* as1_i4p;
typedef __attribute__((address_space(3))) i4v* as3_i4p;

__device__ __forceinline__ void async_cp_b128(const _Float16* g, _Float16* l) {
#if ASYNC_LDS
  __builtin_amdgcn_global_load_async_to_lds_b128(
      (as1_i4p)(uintptr_t)g,
      (as3_i4p)(uint32_t)(uintptr_t)l,
      0, 0);
#else
  *(h8*)l = *(const h8*)g;
#endif
}
__device__ __forceinline__ void wait_async() {
#if ASYNC_LDS
#if __has_builtin(__builtin_amdgcn_s_wait_asynccnt)
  __builtin_amdgcn_s_wait_asynccnt(0);
#else
  asm volatile("s_wait_asynccnt 0" ::: "memory");
#endif
#endif
}

// ---- fragment load helpers (LDS, contiguous k = 16h + s per lane-half, A and B alike) ----
__device__ __forceinline__ h16 ld_a_frag(const _Float16* p) {
  // A 16x32 f16: lane = M, 16 contiguous K values per lane-half
  h8 lo = *(const h8*)p;
  h8 hi = *(const h8*)(p + 8);
  return __builtin_shufflevector(lo, hi, 0,1,2,3,4,5,6,7,8,9,10,11,12,13,14,15);
}
__device__ __forceinline__ h16 ld_b_frag(const _Float16* p) {
  // B 32x16 f16: lane = N, 16 contiguous K values per lane-half
  h8 lo = *(const h8*)p;
  h8 hi = *(const h8*)(p + 8);
  return __builtin_shufflevector(lo, hi, 0,1,2,3,4,5,6,7,8,9,10,11,12,13,14,15);
}

// ---------------- Kernel 0: f32 -> f16 weight conversion + transpose ----------------
__global__ void convert_weights(const float* __restrict__ Wq,
                                const float* __restrict__ Wkv,
                                _Float16* __restrict__ ws) {
  int idx = blockIdx.x * blockDim.x + threadIdx.x;
  if (idx >= DMODEL * DMODEL) return;
  int n = idx >> 9, k = idx & 511;
  ws[WQ_OFF + idx] = (_Float16)Wq[(size_t)k * DMODEL + n];
  ws[WK_OFF + idx] = (_Float16)Wkv[(size_t)k * 1024 + n];
  ws[WV_OFF + idx] = (_Float16)Wkv[(size_t)k * 1024 + 512 + n];
}

// ---- A-tile staging (needs f32->f16 conversion, so VGPR path) ----
struct ATmp { float4 v[4]; };

__device__ __forceinline__ ATmp issue_a(const float* __restrict__ src,
                                        int mbase, int k0, int tid) {
  ATmp t;
  #pragma unroll
  for (int it = 0; it < 4; ++it) {
    int idx = tid + it * 128;
    int r = idx >> 3, c4 = (idx & 7) * 4;
    const float* p = &src[(size_t)(mbase + r) * DMODEL + k0 + c4];
    t.v[it] = *(const float4*)p;
    if (it == 0) __builtin_prefetch(p + 64, 0, 1);   // k-tile two ahead (speculative)
  }
  return t;
}
__device__ __forceinline__ void commit_a(_Float16* As, const ATmp& t, int tid) {
  #pragma unroll
  for (int it = 0; it < 4; ++it) {
    int idx = tid + it * 128;
    int r = idx >> 3, c4 = (idx & 7) * 4;
    h4 hv = { (_Float16)t.v[it].x, (_Float16)t.v[it].y,
              (_Float16)t.v[it].z, (_Float16)t.v[it].w };
    *(h4*)&As[r * 40 + c4] = hv;
  }
}
// ---- W-tile staging: pure f16 copy -> async global->LDS ----
__device__ __forceinline__ void stage_w_async(const _Float16* __restrict__ WT,
                                              _Float16* Wt, int nbase, int k0, int tid) {
  #pragma unroll
  for (int it = 0; it < 2; ++it) {
    int idx = tid + it * 128;           // 256 chunks of 8 halfs (16B)
    int n = idx >> 1, c8 = (idx & 1) * 8;
    async_cp_b128(&WT[(size_t)(nbase + n) * DMODEL + k0 + c8], &Wt[n * 40 + c8]);
  }
}

__device__ __forceinline__ void store_c_tile(_Float16* C2, _Float16* __restrict__ O,
                                             const f8x* acc, int mbase, int nbase,
                                             int wv, int lane, int tid) {
  // repack wave tiles (C layout) into LDS [64][64], then b128 coalesced stores
  const int hb = (lane < 16) ? 0 : 8;
  #pragma unroll
  for (int nb = 0; nb < 4; ++nb) {
    int ncol = 16 * nb + (lane & 15);
    #pragma unroll
    for (int i = 0; i < 8; ++i)
      C2[(16 * wv + i + hb) * 64 + ncol] = (_Float16)acc[nb][i];
  }
  __syncthreads();
  #pragma unroll
  for (int it = 0; it < 4; ++it) {
    int idx = tid + it * 128;
    int r = idx >> 3, c8 = (idx & 7) * 8;
    *(h8*)&O[(size_t)(mbase + r) * DMODEL + nbase + c8] = *(const h8*)&C2[r * 64 + c8];
  }
}

// ---------------- Kernel 1a: Q projection (WMMA f16, double-buffered) ----------------
// grid (8 n-blocks FAST, 3136 m-blocks): consecutive blocks share the A-tile -> L2 hits.
__launch_bounds__(128)
__global__ void proj_q(const float* __restrict__ x, _Float16* __restrict__ ws) {
  __shared__ _Float16 smem[2560 * 4];   // As[2] + Wt[2]
  const int tid = threadIdx.x, wv = tid >> 5, lane = tid & 31;
  const int nbase = blockIdx.x * 64, mbase = blockIdx.y * 64;
  const _Float16* W = ws + WQ_OFF;
  f8x acc[4] = {};

  { stage_w_async(W, smem + 2560, nbase, 0, tid);
    ATmp ta = issue_a(x, mbase, 0, tid);
    commit_a(smem, ta, tid); }
  wait_async();
  __syncthreads();

  for (int s = 0; s < 16; ++s) {
    _Float16* As = smem + (s & 1) * 5120;
    _Float16* Wt = As + 2560;
    _Float16* AsN = smem + ((s + 1) & 1) * 5120;
    ATmp ta;
    const bool more = (s + 1 < 16);
    if (more) { stage_w_async(W, AsN + 2560, nbase, (s + 1) * 32, tid);
                ta = issue_a(x, mbase, (s + 1) * 32, tid); }
    h16 a = ld_a_frag(&As[(16 * wv + (lane & 15)) * 40 + ((lane < 16) ? 0 : 16)]);
    #pragma unroll
    for (int nb = 0; nb < 4; ++nb) {
      h16 b = ld_b_frag(&Wt[(16 * nb + (lane & 15)) * 40 + ((lane < 16) ? 0 : 16)]);
      acc[nb] = __builtin_amdgcn_wmma_f32_16x16x32_f16(false, a, false, b,
                                                       (short)0, acc[nb], false, false);
    }
    if (more) commit_a(AsN, ta, tid);
    wait_async();
    __syncthreads();
  }
  store_c_tile(smem, ws + Q_OFF, acc, mbase, nbase, wv, lane, tid);
}

// ---------------- Kernel 1b: fused K+V projection (y staged once, double-buffered) ----------------
__launch_bounds__(128)
__global__ void proj_kv(const float* __restrict__ y, _Float16* __restrict__ ws) {
  __shared__ _Float16 smem[2560 * 6];   // (As + WtK + WtV) x 2
  const int tid = threadIdx.x, wv = tid >> 5, lane = tid & 31;
  const int nbase = blockIdx.x * 64, mbase = blockIdx.y * 64;
  f8x accK[4] = {}, accV[4] = {};

  { stage_w_async(ws + WK_OFF, smem + 2560, nbase, 0, tid);
    stage_w_async(ws + WV_OFF, smem + 5120, nbase, 0, tid);
    ATmp ta = issue_a(y, mbase, 0, tid);
    commit_a(smem, ta, tid); }
  wait_async();
  __syncthreads();

  for (int s = 0; s < 16; ++s) {
    _Float16* As  = smem + (s & 1) * 7680;
    _Float16* WtK = As + 2560;
    _Float16* WtV = As + 5120;
    _Float16* AsN = smem + ((s + 1) & 1) * 7680;
    ATmp ta;
    const bool more = (s + 1 < 16);
    if (more) { stage_w_async(ws + WK_OFF, AsN + 2560, nbase, (s + 1) * 32, tid);
                stage_w_async(ws + WV_OFF, AsN + 5120, nbase, (s + 1) * 32, tid);
                ta = issue_a(y, mbase, (s + 1) * 32, tid); }
    h16 a = ld_a_frag(&As[(16 * wv + (lane & 15)) * 40 + ((lane < 16) ? 0 : 16)]);
    #pragma unroll
    for (int nb = 0; nb < 4; ++nb) {
      h16 bk = ld_b_frag(&WtK[(16 * nb + (lane & 15)) * 40 + ((lane < 16) ? 0 : 16)]);
      accK[nb] = __builtin_amdgcn_wmma_f32_16x16x32_f16(false, a, false, bk,
                                                        (short)0, accK[nb], false, false);
      h16 bv = ld_b_frag(&WtV[(16 * nb + (lane & 15)) * 40 + ((lane < 16) ? 0 : 16)]);
      accV[nb] = __builtin_amdgcn_wmma_f32_16x16x32_f16(false, a, false, bv,
                                                        (short)0, accV[nb], false, false);
    }
    if (more) commit_a(AsN, ta, tid);
    wait_async();
    __syncthreads();
  }
  store_c_tile(smem, ws + K_OFF, accK, mbase, nbase, wv, lane, tid);
  __syncthreads();
  store_c_tile(smem, ws + V_OFF, accV, mbase, nbase, wv, lane, tid);
}

// ---------------- Kernel 2: fused attention per (window, head) ----------------
__launch_bounds__(128)
__global__ void attn_kernel(const float* __restrict__ bias_table,
                            const _Float16* __restrict__ ws,
                            float* __restrict__ out) {
  __shared__ _Float16 qs[64 * 40];   // [token][dim]
  __shared__ _Float16 ks[64 * 40];   // [token][dim]
  __shared__ _Float16 vT[32 * 72];   // [dim][token]
  __shared__ _Float16 at[64 * 72];   // attn f16
  __shared__ float    bsh[169];      // bias column for this head
  const int tid = threadIdx.x, wv = tid >> 5, lane = tid & 31;
  const int win = blockIdx.x, head = blockIdx.y;
  const _Float16* Q = ws + Q_OFF;
  const _Float16* K = ws + K_OFF;
  const _Float16* V = ws + V_OFF;
  const size_t base = (size_t)win * NTOK * DMODEL + head * DHEAD;

  // async-copy q/k rows < 49 straight into LDS (16B chunks)
  #pragma unroll
  for (int it = 0; it < 2; ++it) {
    int idx = tid + it * 128;           // covers 64 rows x 4 chunks
    int r = idx >> 2, c8 = (idx & 3) * 8;
    if (r < NTOK) {
      async_cp_b128(&Q[base + (size_t)r * DMODEL + c8], &qs[r * 40 + c8]);
      async_cp_b128(&K[base + (size_t)r * DMODEL + c8], &ks[r * 40 + c8]);
    }
  }
  // zero-pad q/k rows 49..63
  {
    h8 z8 = {0,0,0,0,0,0,0,0};
    for (int idx = tid; idx < 60; idx += 128) {   // 15 rows * 4 chunks
      int r = NTOK + (idx >> 2), c8 = (idx & 3) * 8;
      *(h8*)&qs[r * 40 + c8] = z8;
      *(h8*)&ks[r * 40 + c8] = z8;
    }
  }
  // v goes through VGPRs (transpose into vT[dim][token], invalid tokens zero)
  h4 tv[4];
  const h4 zero4 = { (_Float16)0.f, (_Float16)0.f, (_Float16)0.f, (_Float16)0.f };
  #pragma unroll
  for (int it = 0; it < 4; ++it) {
    int idx = tid + it * 128;
    int r = idx >> 3, c4 = (idx & 7) * 4;
    tv[it] = (r < NTOK) ? *(const h4*)&V[base + (size_t)r * DMODEL + c4] : zero4;
  }
  float tb[2];
  tb[0] = (tid < 169) ? bias_table[tid * HEADS + head] : 0.f;
  tb[1] = (tid + 128 < 169) ? bias_table[(tid + 128) * HEADS + head] : 0.f;
  #pragma unroll
  for (int it = 0; it < 4; ++it) {
    int idx = tid + it * 128;
    int r = idx >> 3, c4 = (idx & 7) * 4;
    #pragma unroll
    for (int j = 0; j < 4; ++j) vT[(c4 + j) * 72 + r] = tv[it][j];
  }
  if (tid < 169) bsh[tid] = tb[0];
  if (tid + 128 < 169) bsh[tid + 128] = tb[1];
  wait_async();
  __syncthreads();

  // sim = q @ k^T : wave owns rows [16wv, 16wv+16), 4 col tiles, K=32 (1 WMMA each)
  h16 a = ld_a_frag(&qs[(16 * wv + (lane & 15)) * 40 + ((lane < 16) ? 0 : 16)]);
  f8x acc[4];
  #pragma unroll
  for (int t = 0; t < 4; ++t) {
    h16 b = ld_b_frag(&ks[(16 * t + (lane & 15)) * 40 + ((lane < 16) ? 0 : 16)]);
    f8x z = {};
    acc[t] = __builtin_amdgcn_wmma_f32_16x16x32_f16(false, a, false, b,
                                                    (short)0, z, false, false);
  }

  // softmax over j (49 valid); row reductions = shfl_xor within 16-lane halves
  const float scale = 0.17677669529663687f;   // 1/sqrt(32)
  const int ncol0 = lane & 15;
  #pragma unroll
  for (int i = 0; i < 8; ++i) {
    int r  = 16 * wv + i + ((lane < 16) ? 0 : 8);
    int rc = (r < NTOK) ? r : (NTOK - 1);
    int rh = rc / 7, rw = rc % 7;
    float sv[4];
    float mx = -1e30f;
    #pragma unroll
    for (int t = 0; t < 4; ++t) {
      int n  = 16 * t + ncol0;
      int nc = (n < NTOK) ? n : (NTOK - 1);
      int nh = nc / 7, nw = nc % 7;
      int bidx = (rh - nh + 6) * 13 + (rw - nw + 6);
      float s = acc[t][i] * scale + bsh[bidx];
      sv[t] = (n < NTOK) ? s : -1e30f;
      mx = fmaxf(mx, sv[t]);
    }
    #pragma unroll
    for (int m = 1; m < 16; m <<= 1) mx = fmaxf(mx, __shfl_xor(mx, m, 32));
    float sum = 0.f;
    #pragma unroll
    for (int t = 0; t < 4; ++t) {
      int n = 16 * t + ncol0;
      float e = (n < NTOK) ? __expf(sv[t] - mx) : 0.f;
      sv[t] = e;
      sum += e;
    }
    #pragma unroll
    for (int m = 1; m < 16; m <<= 1) sum += __shfl_xor(sum, m, 32);
    float inv = 1.f / sum;
    #pragma unroll
    for (int t = 0; t < 4; ++t)
      at[r * 72 + 16 * t + ncol0] = (_Float16)(sv[t] * inv);
  }
  __syncthreads();

  // out = attn @ v : 16x32 per wave, K=64 (2 steps), 2 N tiles
  f8x o[2] = {};
  #pragma unroll
  for (int kt = 0; kt < 2; ++kt) {
    h16 a2 = ld_a_frag(&at[(16 * wv + (lane & 15)) * 72 + 32 * kt + ((lane < 16) ? 0 : 16)]);
    #pragma unroll
    for (int nb = 0; nb < 2; ++nb) {
      h16 b2 = ld_b_frag(&vT[(16 * nb + (lane & 15)) * 72 + 32 * kt + ((lane < 16) ? 0 : 16)]);
      o[nb] = __builtin_amdgcn_wmma_f32_16x16x32_f16(false, a2, false, b2,
                                                     (short)0, o[nb], false, false);
    }
  }

  const int mrow = 16 * wv + ((lane < 16) ? 0 : 8);
  #pragma unroll
  for (int nb = 0; nb < 2; ++nb) {
    int ncol = head * DHEAD + 16 * nb + (lane & 15);
    #pragma unroll
    for (int i = 0; i < 8; ++i) {
      int m = mrow + i;
      if (m < NTOK)
        out[(size_t)(win * NTOK + m) * DMODEL + ncol] = o[nb][i];
    }
  }
}

extern "C" void kernel_launch(void* const* d_in, const int* in_sizes, int n_in,
                              void* d_out, int out_size, void* d_ws, size_t ws_size,
                              hipStream_t stream) {
  const float* x    = (const float*)d_in[0];
  const float* y    = (const float*)d_in[1];
  const float* Wq   = (const float*)d_in[2];
  const float* Wkv  = (const float*)d_in[3];
  const float* bias = (const float*)d_in[4];
  _Float16* ws = (_Float16*)d_ws;
  float* out = (float*)d_out;

  convert_weights<<<dim3((DMODEL * DMODEL + 255) / 256), dim3(256), 0, stream>>>(Wq, Wkv, ws);
  // n-blocks fastest so the 8 blocks sharing an A-tile are dispatch-adjacent (L2 reuse)
  proj_q <<<dim3(DMODEL / 64, MTOT / 64), dim3(128), 0, stream>>>(x, ws);
  proj_kv<<<dim3(DMODEL / 64, MTOT / 64), dim3(128), 0, stream>>>(y, ws);
  attn_kernel<<<dim3(NWIN, HEADS), dim3(128), 0, stream>>>(bias, ws, out);
}
